// Qwen3VLMoeTextTopKRouter_82360292868550
// MI455X (gfx1250) — matrix operand order repaired and checked by reading, not compiled, not run on hardware
//
#include <hip/hip_runtime.h>

#define TOKENS   32768
#define HIDDEN   2048
#define NEXP     128
#define TOPK     8
#define BLK_TOK  128   // tokens per workgroup (8 waves x 16)
#define LDS_STRIDE 132 // 128 + 4 pad: bank-conflict-free (132 % 64 == 4)

typedef __attribute__((ext_vector_type(16))) __bf16 v16bf;
typedef __attribute__((ext_vector_type(8)))  float  v8f;

// fp32 -> bf16, round-to-nearest-even (bit-level, no FP traffic)
static __device__ __forceinline__ unsigned short f2bf_bits(float f) {
    unsigned u = __builtin_bit_cast(unsigned, f);
    u += 0x7FFFu + ((u >> 16) & 1u);
    return (unsigned short)(u >> 16);
}
static __device__ __forceinline__ __bf16 f2bf(float f) {
    unsigned short h = f2bf_bits(f);
    return __builtin_bit_cast(__bf16, h);
}

// ---------------------------------------------------------------------------
// Kernel 1: one-time convert router weight [128 x 2048] fp32 -> bf16 in d_ws.
// 512 KB result stays resident in the 192 MB L2 for the whole GEMM.
// ---------------------------------------------------------------------------
__global__ __launch_bounds__(256)
void convert_w_kernel(const float* __restrict__ w, unsigned short* __restrict__ wb) {
    int i = (blockIdx.x * 256 + threadIdx.x) * 4;
    float4 v = *(const float4*)(w + i);
    uint2 packed;
    packed.x = (unsigned)f2bf_bits(v.x) | ((unsigned)f2bf_bits(v.y) << 16);
    packed.y = (unsigned)f2bf_bits(v.z) | ((unsigned)f2bf_bits(v.w) << 16);
    *(uint2*)(wb + i) = packed;
}

// ---------------------------------------------------------------------------
// Kernel 2: fused router. Per block: 128 tokens x 128 experts.
//   Phase 1: bf16 WMMA GEMM (fp32 accum), logits -> LDS.
//   Phase 2: per-token top-8 + softmax-renorm (Z cancels) + dense scatter.
// ---------------------------------------------------------------------------
__global__ __launch_bounds__(256)
void router_kernel(const float* __restrict__ x,
                   const unsigned short* __restrict__ wb,
                   float* __restrict__ scores,
                   int* __restrict__ topidx) {
    __shared__ __align__(16) float s_logits[BLK_TOK * LDS_STRIDE];

    const int t0   = blockIdx.x * BLK_TOK;
    const int lane = threadIdx.x & 31;
    const int wave = threadIdx.x >> 5;

    // ---------------- Phase 1: GEMM ----------------
    v8f acc[8] = {};  // 8 tiles of 16x16 f32: experts 0..127 for 16 tokens

    const int m0    = t0 + wave * 16;
    const int arow  = m0 + (lane & 15);          // token row this lane loads for A
    const int ahalf = (lane >> 4) * 16;          // A K-offset: 0 (lo lanes) / 16 (hi lanes)
    const int bhalf = (lane >> 4) * 16;          // B K-offset: 0 / 16
    const int bexp  = lane & 15;                 // expert column within 16-wide tile
    const float* xrow = x + (size_t)arow * HIDDEN;

    for (int kc = 0; kc < HIDDEN; kc += 32) {
        // A tile: contiguous k order (same as B) — lane m<16: K=0..15; m>=16: K=16..31
        const float4* xp = (const float4*)(xrow + kc + ahalf);
        float4 p0 = xp[0];
        float4 p1 = xp[1];
        const float4* xq = (const float4*)(xrow + kc + ahalf + 8);
        float4 p2 = xq[0];
        float4 p3 = xq[1];

        v16bf a;
        a[0]  = f2bf(p0.x); a[1]  = f2bf(p0.y); a[2]  = f2bf(p0.z); a[3]  = f2bf(p0.w);
        a[4]  = f2bf(p1.x); a[5]  = f2bf(p1.y); a[6]  = f2bf(p1.z); a[7]  = f2bf(p1.w);
        a[8]  = f2bf(p2.x); a[9]  = f2bf(p2.y); a[10] = f2bf(p2.z); a[11] = f2bf(p2.w);
        a[12] = f2bf(p3.x); a[13] = f2bf(p3.y); a[14] = f2bf(p3.z); a[15] = f2bf(p3.w);

        #pragma unroll
        for (int e = 0; e < 8; ++e) {
            // B tile 32x16: lane n<16 -> N=n, K=kc..kc+15; n>=16 -> N=n-16, K=kc+16..kc+31
            const v16bf b = *(const v16bf*)(wb + (size_t)(e * 16 + bexp) * HIDDEN + kc + bhalf);
            acc[e] = __builtin_amdgcn_wmma_f32_16x16x32_bf16(
                false, a, false, b, (short)0, acc[e], false, false);
        }
    }

    // Spill logits to LDS. C layout: VGPR j -> M = j (lanes 0-15) / j+8 (lanes 16-31), N = lane%16
    const int rbase = wave * 16 + ((lane >> 4) * 8);
    const int col0  = lane & 15;
    #pragma unroll
    for (int e = 0; e < 8; ++e) {
        #pragma unroll
        for (int j = 0; j < 8; ++j)
            s_logits[(rbase + j) * LDS_STRIDE + e * 16 + col0] = acc[e][j];
    }

    __syncthreads();

    // ---------------- Phase 2: top-8 + renorm softmax ----------------
    const int t = threadIdx.x;
    if (t < BLK_TOK) {
        float* row = s_logits + t * LDS_STRIDE;

        float topv[TOPK];
        int   topi[TOPK];
        #pragma unroll 1
        for (int k = 0; k < TOPK; ++k) {        // 8 masked argmax passes (descending order)
            float best = -3.402823466e38f;
            int   bi   = 0;
            for (int e = 0; e < NEXP; ++e) {
                float v = row[e];
                if (v > best) { best = v; bi = e; }
            }
            topv[k] = best;
            topi[k] = bi;
            row[bi] = -3.402823466e38f;
        }

        // softmax Z cancels under top-k renorm: p_k = exp(l_k - m) / sum_top8 exp(l_j - m)
        const float mx = topv[0];
        float ev[TOPK], s = 0.0f;
        #pragma unroll
        for (int k = 0; k < TOPK; ++k) { ev[k] = __expf(topv[k] - mx); s += ev[k]; }
        const float inv = 1.0f / s;

        // rebuild dense score row in LDS, then stream out as b128 stores
        for (int e = 0; e < NEXP; ++e) row[e] = 0.0f;
        #pragma unroll
        for (int k = 0; k < TOPK; ++k) row[topi[k]] = ev[k] * inv;

        float4*       orow = (float4*)(scores + (size_t)(t0 + t) * NEXP);
        const float4* rr   = (const float4*)row;   // t*132*4B = t*528B, 16B aligned
        #pragma unroll
        for (int i = 0; i < NEXP / 4; ++i) orow[i] = rr[i];

        int* irow = topidx + (size_t)(t0 + t) * TOPK;
        #pragma unroll
        for (int k = 0; k < TOPK; ++k) irow[k] = topi[k];
    }
}

// ---------------------------------------------------------------------------
extern "C" void kernel_launch(void* const* d_in, const int* in_sizes, int n_in,
                              void* d_out, int out_size, void* d_ws, size_t ws_size,
                              hipStream_t stream) {
    const float* x  = (const float*)d_in[0];               // [TOKENS, HIDDEN] fp32
    const float* wt = (const float*)d_in[1];               // [NEXP,   HIDDEN] fp32
    float* scores   = (float*)d_out;                       // [TOKENS, NEXP] fp32
    int*   topidx   = (int*)(scores + (size_t)TOKENS * NEXP); // [TOKENS, TOPK] int32
    unsigned short* wb = (unsigned short*)d_ws;            // bf16 W, 512 KB scratch

    convert_w_kernel<<<(NEXP * HIDDEN) / (256 * 4), 256, 0, stream>>>(wt, wb);
    router_kernel<<<TOKENS / BLK_TOK, 256, 0, stream>>>(x, wb, scores, topidx);
}
